// AddModelWithAttentionStacked_80187039416904
// MI455X (gfx1250) — compile-verified, matrix operand order checked by reading, not run on hardware
//
#include <hip/hip_runtime.h>

typedef __bf16 bf16;
typedef __attribute__((ext_vector_type(16))) __bf16 v16bf;
typedef __attribute__((ext_vector_type(8)))  float  v8f;
typedef __attribute__((ext_vector_type(4)))  unsigned int v4u;
typedef __attribute__((ext_vector_type(8)))  int v8i;
typedef __attribute__((ext_vector_type(4)))  int v4i;

#define DEV static __device__ __forceinline__

#if __has_builtin(__builtin_amdgcn_tensor_load_to_lds)
#define HAVE_TDM 1
#else
#define HAVE_TDM 0
#endif

// ---------------- model dims ----------------
#define GV   32000
#define EV   256
#define KH   8
#define DL   6
#define BV   4
#define LV   512
#define MV   64
#define KNV  4
#define KE   2048          // KH*EV
#define BL   (BV*LV)       // 2048
#define NROWS (BV*MV*KNV)  // 1024 head rows
#define NPARTS 25          // G split: 25 * 80 tiles * 16 = 32000
#define GTILES_PER_PART 80

// ---------------- WMMA fragment helpers (16x16x32 bf16) ----------------
// 16-bit A-matrix 16x32 layout (ISA 7.12.2): lane = {half, idx}, vgpr v holds K pair:
//   K = ((v&4)<<2) + half*8 + (v&3)*2 + (e&1)
DEV int frag_k(int e, int half) {
  int v = e >> 1;
  return ((v & 4) << 2) + (half << 3) + ((v & 3) << 1) + (e & 1);
}

DEV v16bf load_a_frag(const bf16* __restrict__ A, long lda, int lane) {
  v16bf a;
  int row = lane & 15, half = lane >> 4;
#pragma unroll
  for (int e = 0; e < 16; ++e)
    a[e] = A[(long)row * lda + frag_k(e, half)];
  return a;
}

// B stored K x N row-major
DEV v16bf load_b_fragN(const bf16* __restrict__ B, long ldb, int lane) {
  v16bf b;
  int col = lane & 15, half = lane >> 4;
#pragma unroll
  for (int e = 0; e < 16; ++e)
    b[e] = B[(long)frag_k(e, half) * ldb + col];
  return b;
}

// B stored N x K row-major, used as (K x N)  (i.e. multiply by B^T)
DEV v16bf load_b_fragT(const bf16* __restrict__ B, long ldb, int lane) {
  v16bf b;
  int col = lane & 15, half = lane >> 4;
#pragma unroll
  for (int e = 0; e < 16; ++e)
    b[e] = B[(long)col * ldb + frag_k(e, half)];
  return b;
}

DEV v8f wmma_bf16(v16bf a, v16bf b, v8f c) {
  return __builtin_amdgcn_wmma_f32_16x16x32_bf16(false, a, false, b, (short)0, c,
                                                 false, false);
}

// ---------------- generic batched WMMA GEMM ----------------
// C[z] = alpha * A[z] @ op(B[z]) (+ C_prev) (+bias) (relu?)
// one wave == one 16x16 output tile; grid = (N/16, M/16, batches)
template <int TRANSB>
__global__ __launch_bounds__(32)
void k_gemm(const bf16* __restrict__ A, const bf16* __restrict__ B,
            float* Cf, bf16* Cb, const float* __restrict__ bias,
            int Kdim, long lda, long ldb, long ldc,
            long sA0, long sA1, long sB0, long sB1, long sC0, long sC1,
            int nsub, float alpha, int addC, int relu) {
  int lane = threadIdx.x;
  int nb = blockIdx.x, mb = blockIdx.y, z = blockIdx.z;
  int z0 = z / nsub, z1 = z - z0 * nsub;

  const bf16* Ap = A + (long)z0 * sA0 + (long)z1 * sA1 + (long)(mb * 16) * lda;
  const bf16* Bp;
  if (TRANSB)
    Bp = B + (long)z0 * sB0 + (long)z1 * sB1 + (long)(nb * 16) * ldb;
  else
    Bp = B + (long)z0 * sB0 + (long)z1 * sB1 + (long)(nb * 16);

  v8f c = {0.f, 0.f, 0.f, 0.f, 0.f, 0.f, 0.f, 0.f};
  for (int k = 0; k < Kdim; k += 32) {
    v16bf a = load_a_frag(Ap + k, lda, lane);
    v16bf b;
    if (TRANSB)
      b = load_b_fragT(Bp + k, ldb, lane);
    else
      b = load_b_fragN(Bp + (long)k * ldb, ldb, lane);
    c = wmma_bf16(a, b, c);
  }

  int col = nb * 16 + (lane & 15);
  int half = lane >> 4;
  long cbase = (long)z0 * sC0 + (long)z1 * sC1;
  float bcol = bias ? bias[col] : 0.f;
#pragma unroll
  for (int i = 0; i < 8; ++i) {
    int row = mb * 16 + i + half * 8;
    long idx = cbase + (long)row * ldc + col;
    float v = c[i] * alpha + bcol;
    if (addC) v += Cf[idx];
    if (relu) v = v > 0.f ? v : 0.f;
    if (Cf) Cf[idx] = v;
    if (Cb) Cb[idx] = (bf16)v;
  }
}

// ---------------- LDS-staged GEMM with async-copy (A) + TDM (B):  C = [relu](A @ B^T)
// A: (M x K) row-major, B: (N x K) row-major, C: bf16 (M x N).
// Block = 256 threads = 8 waves; block tile 64(M) x 32(N); K-step 64.
// A panel staged via GLOBAL_LOAD_ASYNC_TO_LDS (ASYNCcnt); B panel staged via
// the Tensor Data Mover (TENSOR_LOAD_TO_LDS, TENSORcnt) issued by wave 0.
__global__ __launch_bounds__(256)
void k_gemm_lds(const bf16* __restrict__ A, const bf16* __restrict__ B,
                bf16* __restrict__ Cb, int Kdim, long lda, long ldb, long ldc,
                int relu) {
  __shared__ bf16 As[2][64 * 64];  // 8KB per buffer
  __shared__ bf16 Bs[2][32 * 64];  // 4KB per buffer  (TDM tile: 32 rows x 64 k)

  int tid = threadIdx.x;
  int lane = tid & 31, wave = tid >> 5;
  int wr = wave >> 1, wc = wave & 1;  // wave tile: rows wr*16, cols wc*16
  int mblk = blockIdx.y * 64, nblk = blockIdx.x * 32;

  // A panel 64 rows x 64 k: thread t -> row t>>2, k = (t&3)*16 (+8), 2x b128
  int ar = tid >> 2, ak = (tid & 3) * 16;
  const bf16* Ag = A + (long)(mblk + ar) * lda + ak;

  unsigned lA0[2], lA1[2], lB[2];
#pragma unroll
  for (int p = 0; p < 2; ++p) {
    lA0[p] = (unsigned)(unsigned long)&As[p][ar * 64 + ak];
    lA1[p] = (unsigned)(unsigned long)&As[p][ar * 64 + ak + 8];
    lB[p]  = (unsigned)(unsigned long)&Bs[p][0];
  }

  auto issueA = [&](int p, int k0) {
    unsigned long ga0 = (unsigned long)(Ag + k0);
    unsigned long ga1 = (unsigned long)(Ag + k0 + 8);
    asm volatile("global_load_async_to_lds_b128 %0, %1, off"
                 :: "v"(lA0[p]), "v"(ga0) : "memory");
    asm volatile("global_load_async_to_lds_b128 %0, %1, off"
                 :: "v"(lA1[p]), "v"(ga1) : "memory");
  };

#if HAVE_TDM
  // TDM descriptor: 2D tile, tile_dim0 = 64 k-elements, tile_dim1 = 32 rows of B,
  // dim0 stride = ldb elements, data_size = 2 bytes. LDS layout == Bs[row][k].
  auto issueB = [&](int p, int k0) {
    if (wave == 0) {
      unsigned long gb = (unsigned long)(B + (long)nblk * ldb + k0);
      v4u g0;
      g0[0] = 1u;                          // count=1, user descriptor
      g0[1] = lB[p];                       // lds_addr
      g0[2] = (unsigned)gb;                // global_addr[31:0]
      g0[3] = (unsigned)((gb >> 32) & 0x1FFFFFFu) | 0x80000000u;  // addr[56:32], type=2
      int td0 = Kdim;                      // tensor_dim0 (OOB guard)
      int td1 = 1 << 20;                   // tensor_dim1 (OOB guard)
      unsigned long s0 = (unsigned long)ldb;
      v8i g1;
      g1[0] = 0x10000;                                   // data_size=1 (2 bytes)
      g1[1] = (int)((td0 & 0xFFFF) << 16);               // tensor_dim0 lo
      g1[2] = (int)(((td0 >> 16) & 0xFFFF) | ((td1 & 0xFFFF) << 16));
      g1[3] = (int)(((td1 >> 16) & 0xFFFF) | (64 << 16));  // tile_dim0 = 64
      g1[4] = 32;                                          // tile_dim1 = 32
      g1[5] = (int)(s0 & 0xFFFFFFFFu);                     // dim0_stride lo
      g1[6] = (int)((s0 >> 32) & 0xFFFFu);                 // dim0_stride hi
      g1[7] = 0;
      v4i g2 = {0, 0, 0, 0};
      v4i g3 = {0, 0, 0, 0};
#if __clang_major__ >= 23
      v8i g4 = {0, 0, 0, 0, 0, 0, 0, 0};
      __builtin_amdgcn_tensor_load_to_lds(g0, g1, g2, g3, g4, 0);
#else
      __builtin_amdgcn_tensor_load_to_lds(g0, g1, g2, g3, 0);
#endif
    }
  };
#else
  // fallback: async-copy B panel as well (1x b128 per thread)
  int bc = tid >> 3, bk = (tid & 7) * 8;
  const bf16* Bg = B + (long)(nblk + bc) * ldb + bk;
  unsigned lBa[2];
#pragma unroll
  for (int p = 0; p < 2; ++p)
    lBa[p] = (unsigned)(unsigned long)&Bs[p][bc * 64 + bk];
  auto issueB = [&](int p, int k0) {
    unsigned long gb = (unsigned long)(Bg + k0);
    asm volatile("global_load_async_to_lds_b128 %0, %1, off"
                 :: "v"(lBa[p]), "v"(gb) : "memory");
  };
#endif

  int nk = Kdim >> 6;  // K-steps of 64
  issueA(0, 0);
  issueB(0, 0);

  v8f c = {0.f, 0.f, 0.f, 0.f, 0.f, 0.f, 0.f, 0.f};
  int row = wr * 16 + (lane & 15);
  int col = wc * 16 + (lane & 15);
  int half = lane >> 4;

  for (int ks = 0; ks < nk; ++ks) {
    int cur = ks & 1;
    if (ks + 1 < nk) {
      issueA(cur ^ 1, (ks + 1) * 64);
      issueB(cur ^ 1, (ks + 1) * 64);
#if HAVE_TDM
      __builtin_amdgcn_s_wait_tensorcnt(1);  // current TDM tile landed (in-order)
      asm volatile("s_wait_asynccnt 0x2" ::: "memory");
#else
      asm volatile("s_wait_asynccnt 0x3" ::: "memory");
#endif
    } else {
#if HAVE_TDM
      __builtin_amdgcn_s_wait_tensorcnt(0);
      asm volatile("s_wait_asynccnt 0x0" ::: "memory");
#else
      asm volatile("s_wait_asynccnt 0x0" ::: "memory");
#endif
    }
    __syncthreads();
#pragma unroll
    for (int ksub = 0; ksub < 64; ksub += 32) {
      v16bf a, b;
#pragma unroll
      for (int e = 0; e < 16; ++e)
        a[e] = As[cur][row * 64 + ksub + frag_k(e, half)];
#pragma unroll
      for (int e = 0; e < 16; ++e)
        b[e] = Bs[cur][col * 64 + ksub + frag_k(e, half)];
      c = wmma_bf16(a, b, c);
    }
    __syncthreads();  // protect buffer reuse by the next issue
  }

  int gcol = nblk + col;
#pragma unroll
  for (int i = 0; i < 8; ++i) {
    int grow = mblk + wr * 16 + i + half * 8;
    float v = c[i];
    if (relu) v = v > 0.f ? v : 0.f;
    Cb[(long)grow * ldc + gcol] = (bf16)v;
  }
}

// ---------------- elementwise / reduction kernels ----------------
__global__ __launch_bounds__(256)
void k_conv_bf16(const float* __restrict__ in, bf16* __restrict__ out, long n) {
  long i = (long)blockIdx.x * 256 + threadIdx.x;
  if (i < n) out[i] = (bf16)in[i];
}

// xsa = norm(embed[masked] + pos);  one block per (b,l) row of 256
__global__ __launch_bounds__(256)
void k_init(const int* __restrict__ masked, const float* __restrict__ embed,
            const float* __restrict__ pos, float* __restrict__ xf,
            bf16* __restrict__ xb) {
  __shared__ float s1[256], s2[256];
  int r = blockIdx.x, e = threadIdx.x;
  int l = r % LV;
  int g = masked[r];
  float x = embed[(long)g * EV + e] + pos[(long)l * EV + e];
  s1[e] = x; s2[e] = x * x;
  __syncthreads();
  for (int off = 128; off > 0; off >>= 1) {
    if (e < off) { s1[e] += s1[e + off]; s2[e] += s2[e + off]; }
    __syncthreads();
  }
  float mu = s1[0] / 256.f;
  float var = (s2[0] - 256.f * mu * mu) / 255.f;
  float y = x / (1.f + sqrtf(fmaxf(var, 0.f)));
  long idx = (long)r * EV + e;
  xf[idx] = y; xb[idx] = (bf16)y;
}

// rollA[l] = x[l-1], rollB[l] = x[l+1] (cyclic within each b)
__global__ __launch_bounds__(256)
void k_roll(const bf16* __restrict__ xb, bf16* __restrict__ ra,
            bf16* __restrict__ rb) {
  int r = blockIdx.x, e = threadIdx.x;
  int b = r / LV, l = r % LV;
  long d = (long)r * EV + e;
  ra[d] = xb[((long)(b * LV + (l + LV - 1) % LV)) * EV + e];
  rb[d] = xb[((long)(b * LV + (l + 1) % LV)) * EV + e];
}

// xsad = norm(xsad); xsa = norm(xsa + STEP*xsad)
__global__ __launch_bounds__(256)
void k_update(const float* __restrict__ xsad, float* __restrict__ xf,
              bf16* __restrict__ xb) {
  __shared__ float s1[256], s2[256];
  int r = blockIdx.x, e = threadIdx.x;
  long idx = (long)r * EV + e;
  float x = xsad[idx];
  s1[e] = x; s2[e] = x * x;
  __syncthreads();
  for (int off = 128; off > 0; off >>= 1) {
    if (e < off) { s1[e] += s1[e + off]; s2[e] += s2[e + off]; }
    __syncthreads();
  }
  float mu = s1[0] / 256.f;
  float var = (s2[0] - 256.f * mu * mu) / 255.f;
  float xn = x / (1.f + sqrtf(fmaxf(var, 0.f)));
  float t = xf[idx] + 0.05f * xn;
  __syncthreads();
  s1[e] = t; s2[e] = t * t;
  __syncthreads();
  for (int off = 128; off > 0; off >>= 1) {
    if (e < off) { s1[e] += s1[e + off]; s2[e] += s2[e + off]; }
    __syncthreads();
  }
  mu = s1[0] / 256.f;
  var = (s2[0] - 256.f * mu * mu) / 255.f;
  float y = t / (1.f + sqrtf(fmaxf(var, 0.f)));
  xf[idx] = y; xb[idx] = (bf16)y;
}

// softmax over rows of length 512 -> bf16 probabilities
__global__ __launch_bounds__(256)
void k_softmax(const float* __restrict__ S, bf16* __restrict__ P) {
  __shared__ float red[256];
  long r = blockIdx.x; int t = threadIdx.x;
  float a = S[r * 512 + t], b = S[r * 512 + 256 + t];
  red[t] = fmaxf(a, b);
  __syncthreads();
  for (int off = 128; off > 0; off >>= 1) {
    if (t < off) red[t] = fmaxf(red[t], red[t + off]);
    __syncthreads();
  }
  float mx = red[0];
  __syncthreads();
  float ea = __expf(a - mx), eb = __expf(b - mx);
  red[t] = ea + eb;
  __syncthreads();
  for (int off = 128; off > 0; off >>= 1) {
    if (t < off) red[t] += red[t + off];
    __syncthreads();
  }
  float inv = 1.f / red[0];
  P[r * 512 + t] = (bf16)(ea * inv);
  P[r * 512 + 256 + t] = (bf16)(eb * inv);
}

// lptok[b,m,:] = xsa_bf[b, mask[b,m], :]
__global__ __launch_bounds__(256)
void k_gather(const int* __restrict__ mask, const bf16* __restrict__ xb,
              bf16* __restrict__ lptok) {
  int r = blockIdx.x, e = threadIdx.x;
  int b = r >> 6;
  int pos = mask[r];
  lptok[(long)r * EV + e] = xb[((long)(b * LV + pos)) * EV + e];
}

// Fused logits tile + streaming logsumexp over a G-slice.
// grid = (NPARTS, NROWS/16), wave per block. A = xxw (1024 x 256) bf16,
// B = embed (G x 256) bf16 used transposed. Never materializes logits.
__global__ __launch_bounds__(32)
void k_lse_partial(const bf16* __restrict__ xxw, const bf16* __restrict__ emb,
                   float* __restrict__ pmax, float* __restrict__ psum) {
  int lane = threadIdx.x;
  int part = blockIdx.x, rg = blockIdx.y;
  const bf16* Ap = xxw + (long)(rg * 16) * EV;

  v16bf afr[8];
#pragma unroll
  for (int kk = 0; kk < 8; ++kk) afr[kk] = load_a_frag(Ap + kk * 32, EV, lane);

  float rmax[8], rsum[8];
#pragma unroll
  for (int i = 0; i < 8; ++i) { rmax[i] = -1e30f; rsum[i] = 0.f; }

  int g0 = part * GTILES_PER_PART;
  for (int t = 0; t < GTILES_PER_PART; ++t) {
    const bf16* Bp = emb + (long)((g0 + t) * 16) * EV;
    v8f c = {0.f, 0.f, 0.f, 0.f, 0.f, 0.f, 0.f, 0.f};
#pragma unroll
    for (int kk = 0; kk < 8; ++kk) {
      v16bf b = load_b_fragT(Bp + kk * 32, EV, lane);
      c = wmma_bf16(afr[kk], b, c);
    }
#pragma unroll
    for (int i = 0; i < 8; ++i) {
      float v = c[i];
      float m = v;
      for (int d = 1; d < 16; d <<= 1) m = fmaxf(m, __shfl_xor(m, d, 32));
      float mn = fmaxf(rmax[i], m);
      float e = __expf(v - mn);
      for (int d = 1; d < 16; d <<= 1) e += __shfl_xor(e, d, 32);
      rsum[i] = rsum[i] * __expf(rmax[i] - mn) + e;
      rmax[i] = mn;
    }
  }
  if ((lane & 15) == 0) {
    int rbase = rg * 16 + ((lane >> 4) << 3);
#pragma unroll
    for (int i = 0; i < 8; ++i) {
      pmax[(long)(rbase + i) * NPARTS + part] = rmax[i];
      psum[(long)(rbase + i) * NPARTS + part] = rsum[i];
    }
  }
}

__global__ __launch_bounds__(32)
void k_lse_reduce(const float* __restrict__ pmax, const float* __restrict__ psum,
                  float* __restrict__ lse) {
  int r = blockIdx.x, lane = threadIdx.x;
  float pm = lane < NPARTS ? pmax[(long)r * NPARTS + lane] : -1e30f;
  float m = pm;
  for (int d = 16; d; d >>= 1) m = fmaxf(m, __shfl_xor(m, d, 32));
  float s = lane < NPARTS ? psum[(long)r * NPARTS + lane] * __expf(pm - m) : 0.f;
  for (int d = 16; d; d >>= 1) s += __shfl_xor(s, d, 32);
  if (lane == 0) lse[r] = m + logf(s);
}

// logit at target vocab id for each head row n = ((b*M+m)*KN + k)
__global__ __launch_bounds__(32)
void k_ltgt(const bf16* __restrict__ xxw, const float* __restrict__ embed,
            const int* __restrict__ mask, const int* __restrict__ unmasked,
            float* __restrict__ lt) {
  int n = blockIdx.x, lane = threadIdx.x;
  int b = n >> 8, rr = n & 255, m = rr >> 2;
  int pos = mask[b * MV + m];
  int g = unmasked[b * LV + pos];
  float acc = 0.f;
  for (int e = lane; e < EV; e += 32)
    acc += (float)xxw[(long)n * EV + e] * embed[(long)g * EV + e];
  for (int d = 16; d; d >>= 1) acc += __shfl_xor(acc, d, 32);
  if (lane == 0) lt[n] = acc;
}

__global__ __launch_bounds__(256)
void k_loss(const float* __restrict__ lt, const float* __restrict__ lse,
            const float* __restrict__ summer, float* __restrict__ out) {
  __shared__ float sc[256], sw[256];
  int t = threadIdx.x;
  int b = t >> 6, m = t & 63;
  int base = (b * MV + m) * KNV;
  float mx = -1e30f;
#pragma unroll
  for (int k = 0; k < KNV; ++k) mx = fmaxf(mx, lt[base + k] - lse[base + k]);
  float s = 0.f;
#pragma unroll
  for (int k = 0; k < KNV; ++k) s += __expf(lt[base + k] - lse[base + k] - mx);
  float cent = mx + logf(s) - logf((float)KNV);
  float w = summer[b * MV + m];
  sc[t] = cent * w; sw[t] = w;
  __syncthreads();
  for (int off = 32; off > 0; off >>= 1) {
    if (m < off) { sc[t] += sc[t + off]; sw[t] += sw[t + off]; }
    __syncthreads();
  }
  if (m == 0) out[b] = -sc[t] / fmaxf(sw[t], 1.f);
}

// ---------------- host side ----------------
static void gemm(hipStream_t s, bool transb, const bf16* A, const bf16* B,
                 float* Cf, bf16* Cb, const float* bias, int M, int N, int Kdim,
                 long lda, long ldb, long ldc, int b0, int nsub,
                 long sA0, long sA1, long sB0, long sB1, long sC0, long sC1,
                 float alpha, int addC, int relu) {
  dim3 grid(N / 16, M / 16, b0 * nsub);
  if (transb)
    k_gemm<1><<<grid, 32, 0, s>>>(A, B, Cf, Cb, bias, Kdim, lda, ldb, ldc,
                                  sA0, sA1, sB0, sB1, sC0, sC1, nsub, alpha,
                                  addC, relu);
  else
    k_gemm<0><<<grid, 32, 0, s>>>(A, B, Cf, Cb, bias, Kdim, lda, ldb, ldc,
                                  sA0, sA1, sB0, sB1, sC0, sC1, nsub, alpha,
                                  addC, relu);
}

static void gemm_lds(hipStream_t s, const bf16* A, const bf16* B, bf16* Cb,
                     int M, int N, int Kdim, long lda, long ldb, long ldc,
                     int relu) {
  dim3 grid(N / 32, M / 64);
  k_gemm_lds<<<grid, 256, 0, s>>>(A, B, Cb, Kdim, lda, ldb, ldc, relu);
}

extern "C" void kernel_launch(void* const* d_in, const int* in_sizes, int n_in,
                              void* d_out, int out_size, void* d_ws,
                              size_t ws_size, hipStream_t stream) {
  (void)in_sizes; (void)n_in; (void)out_size; (void)ws_size;
  const int*   masked   = (const int*)d_in[0];
  const int*   unmasked = (const int*)d_in[1];
  const int*   maskp    = (const int*)d_in[2];
  const float* summer   = (const float*)d_in[3];
  const float* embed    = (const float*)d_in[4];
  const float* pos      = (const float*)d_in[5];
  const float* Wt       = (const float*)d_in[6];
  const float* Wc       = (const float*)d_in[7];
  const float* Wq       = (const float*)d_in[8];
  const float* Wd       = (const float*)d_in[9];
  const float* Wo       = (const float*)d_in[10];
  const float* Wkc      = (const float*)d_in[11];
  const float* bkc      = (const float*)d_in[12];
  const float* Wem      = (const float*)d_in[13];
  float* out = (float*)d_out;

  char* base = (char*)d_ws;
  size_t off = 0;
  auto alloc = [&](size_t bytes) -> void* {
    void* p = base + off;
    off = (off + bytes + 255) & ~(size_t)255;
    return p;
  };

  // bf16 weight copies
  bf16* eb   = (bf16*)alloc((size_t)GV * EV * 2);
  bf16* pb   = (bf16*)alloc((size_t)LV * EV * 2);
  bf16* wtb  = (bf16*)alloc((size_t)DL * EV * EV * 2);
  bf16* wcb  = (bf16*)alloc((size_t)DL * EV * EV * 2);
  bf16* wqb  = (bf16*)alloc((size_t)DL * KE * EV * 2);
  bf16* wdb  = (bf16*)alloc((size_t)DL * KE * KE * 2);
  bf16* wob  = (bf16*)alloc((size_t)DL * KE * EV * 2);
  bf16* wkcb = (bf16*)alloc((size_t)(EV * KNV) * EV * 2);
  bf16* wemb = (bf16*)alloc((size_t)EV * EV * 2);
  // activations
  float* xf    = (float*)alloc((size_t)BL * EV * 4);
  bf16*  xb    = (bf16*)alloc((size_t)BL * EV * 2);
  float* xsad  = (float*)alloc((size_t)BL * EV * 4);
  bf16*  ra    = (bf16*)alloc((size_t)BL * EV * 2);
  bf16*  rb    = (bf16*)alloc((size_t)BL * EV * 2);
  bf16*  tb    = (bf16*)alloc((size_t)BL * EV * 2);
  bf16*  qb    = (bf16*)alloc((size_t)BL * KE * 2);
  float* Sf    = (float*)alloc((size_t)BV * KH * LV * LV * 4);
  bf16*  Pb    = (bf16*)alloc((size_t)BV * KH * LV * LV * 2);
  bf16*  xidb  = (bf16*)alloc((size_t)BL * KE * 2);
  bf16*  xab   = (bf16*)alloc((size_t)BL * KE * 2);
  bf16*  lptb  = (bf16*)alloc((size_t)BV * MV * EV * 2);
  bf16*  xx0b  = (bf16*)alloc((size_t)BV * MV * EV * KNV * 2);
  bf16*  xx2b  = (bf16*)alloc((size_t)BV * 256 * LV * 2);
  bf16*  xx3b  = (bf16*)alloc((size_t)BV * 256 * EV * 2);
  bf16*  xxwb  = (bf16*)alloc((size_t)BV * 256 * EV * 2);
  float* pmax  = (float*)alloc((size_t)NROWS * NPARTS * 4);
  float* psum  = (float*)alloc((size_t)NROWS * NPARTS * 4);
  float* lsev  = (float*)alloc((size_t)NROWS * 4);
  float* ltv   = (float*)alloc((size_t)NROWS * 4);

  auto conv = [&](const float* in, bf16* o, long n) {
    k_conv_bf16<<<dim3((unsigned)((n + 255) / 256)), 256, 0, stream>>>(in, o, n);
  };
  conv(embed, eb, (long)GV * EV);
  conv(pos, pb, (long)LV * EV);
  conv(Wt, wtb, (long)DL * EV * EV);
  conv(Wc, wcb, (long)DL * EV * EV);
  conv(Wq, wqb, (long)DL * KE * EV);
  conv(Wd, wdb, (long)DL * KE * KE);
  conv(Wo, wob, (long)DL * KE * EV);
  conv(Wkc, wkcb, (long)(EV * KNV) * EV);
  conv(Wem, wemb, (long)EV * EV);

  k_init<<<BL, 256, 0, stream>>>(masked, embed, pos, xf, xb);

  const float inv_sqrt_e = 0.0625f;  // 1/sqrt(256)
  for (int d = 0; d < DL; ++d) {
    const bf16* wt_d = wtb + (long)d * EV * EV;
    const bf16* wc_d = wcb + (long)d * EV * EV;
    const bf16* wq_d = wqb + (long)d * KE * EV;
    const bf16* wd_d = wdb + (long)d * KE * KE;
    const bf16* wo_d = wob + (long)d * KE * EV;

    k_roll<<<BL, 256, 0, stream>>>(xb, ra, rb);
    // t = relu(roll(+1) @ Wt)
    gemm(stream, false, ra, wt_d, nullptr, tb, nullptr, BL, EV, EV,
         EV, EV, EV, 1, 1, 0, 0, 0, 0, 0, 0, 1.f, 0, 1);
    // xsad = t @ Wc
    gemm(stream, false, tb, wc_d, xsad, nullptr, nullptr, BL, EV, EV,
         EV, EV, EV, 1, 1, 0, 0, 0, 0, 0, 0, 1.f, 0, 0);
    // t = relu(roll(-1) @ Wc^T)
    gemm(stream, true, rb, wc_d, nullptr, tb, nullptr, BL, EV, EV,
         EV, EV, EV, 1, 1, 0, 0, 0, 0, 0, 0, 1.f, 0, 1);
    // xsad += t @ Wt^T
    gemm(stream, true, tb, wt_d, xsad, nullptr, nullptr, BL, EV, EV,
         EV, EV, EV, 1, 1, 0, 0, 0, 0, 0, 0, 1.f, 1, 0);
    // q = xsa @ Wq^T  (B,L,K*E)  -- async/TDM LDS-staged
    gemm_lds(stream, xb, wq_d, qb, BL, KE, EV, EV, EV, KE, 0);
    // scores S[b,k] = q[b,:,k,:] @ xsa_b^T * inv_sqrt_e
    gemm(stream, true, qb, xb, Sf, nullptr, nullptr, LV, LV, EV,
         KE, EV, LV, BV, KH,
         (long)LV * KE, EV, (long)LV * EV, 0,
         (long)KH * LV * LV, (long)LV * LV, inv_sqrt_e, 0, 0);
    k_softmax<<<BV * KH * LV, 256, 0, stream>>>(Sf, Pb);
    // xid[b,:,k,:] = P[b,k] @ xsa_b
    gemm(stream, false, Pb, xb, nullptr, xidb, nullptr, LV, EV, LV,
         LV, EV, KE, BV, KH,
         (long)KH * LV * LV, (long)LV * LV, (long)LV * EV, 0,
         (long)LV * KE, EV, 1.f, 0, 0);
    // xa = relu(xid @ Wd^T)  -- async/TDM LDS-staged (dominant GEMM)
    gemm_lds(stream, xidb, wd_d, xab, BL, KE, KE, KE, KE, KE, 1);
    // xsad += xa @ Wo
    gemm(stream, false, xab, wo_d, xsad, nullptr, nullptr, BL, EV, KE,
         KE, EV, EV, 1, 1, 0, 0, 0, 0, 0, 0, 1.f, 1, 0);
    k_update<<<BL, 256, 0, stream>>>(xsad, xf, xb);
  }

  // ---- head ----
  k_gather<<<BV * MV, 256, 0, stream>>>(maskp, xb, lptb);
  // xx0 = lptok @ Wkc^T + bkc   -> viewed as (B, M*KN, E)
  gemm(stream, true, lptb, wkcb, nullptr, xx0b, bkc, BV * MV, EV * KNV, EV,
       EV, EV, EV * KNV, 1, 1, 0, 0, 0, 0, 0, 0, 1.f, 0, 0);
  // xx2[b] = xx0[b] @ xsa_b^T   (256 x 512)
  gemm(stream, true, xx0b, xb, nullptr, xx2b, nullptr, 256, LV, EV,
       EV, EV, LV, BV, 1,
       (long)256 * EV, 0, (long)LV * EV, 0, (long)256 * LV, 0, 1.f, 0, 0);
  // xx3[b] = xx2[b] @ xsa_b     (256 x 256)
  gemm(stream, false, xx2b, xb, nullptr, xx3b, nullptr, 256, EV, LV,
       LV, EV, EV, BV, 1,
       (long)256 * LV, 0, (long)LV * EV, 0, (long)256 * EV, 0, 1.f, 0, 0);
  // xxw[b] = xx3[b] @ Wem
  gemm(stream, false, xx3b, wemb, nullptr, xxwb, nullptr, 256, EV, EV,
       EV, EV, EV, BV, 1,
       (long)256 * EV, 0, 0, 0, (long)256 * EV, 0, 1.f, 0, 0);

  // fused logits + streaming logsumexp over G
  k_lse_partial<<<dim3(NPARTS, NROWS / 16), 32, 0, stream>>>(xxwb, eb, pmax, psum);
  k_lse_reduce<<<NROWS, 32, 0, stream>>>(pmax, psum, lsev);
  k_ltgt<<<NROWS, 32, 0, stream>>>(xxwb, embed, maskp, unmasked, ltv);
  k_loss<<<1, 256, 0, stream>>>(ltv, lsev, summer, out);
}
